// DDCModel_37082747633719
// MI455X (gfx1250) — matrix operand order repaired and checked by reading, not compiled, not run on hardware
//
#include <hip/hip_runtime.h>
#include <hip/hip_bf16.h>
#include <stdint.h>

// Problem constants (match reference)
#define NDIM      4096
#define HORIZON   1024
#define N_ACTIONS 4

// Tiling
#define TILE_ROWS 16                 // WMMA N dimension = output rows per tile
#define KSPLIT    8                  // waves per block, each covers NDIM/KSPLIT of K
#define BLOCK_THREADS (32 * KSPLIT)  // 256 threads = 8 wave32
#define NBLOCKS   (NDIM / TILE_ROWS) // 256 blocks -> 2048 waves per step
#define KCHUNK    (NDIM / KSPLIT)    // 512 -> 16 WMMA iterations per wave

typedef __attribute__((ext_vector_type(16))) __bf16 v16bf;
typedef __attribute__((ext_vector_type(8)))  float  v8f;

union ABReg {
  v16bf v;
  uint4 q[2];
  unsigned short us[16];
};

__device__ __forceinline__ unsigned short f2bf(float f) {
  // round-to-nearest-even f32 -> bf16
  unsigned u = __float_as_uint(f);
  u += 0x7fffu + ((u >> 16) & 1u);
  return (unsigned short)(u >> 16);
}

// ---------------------------------------------------------------------------
// One-time T_mats fp32 -> bf16 conversion (all 4 matrices become L2-resident:
// 128 MiB < 192 MB global L2, and per-step streaming traffic is halved)
// ---------------------------------------------------------------------------
__global__ __launch_bounds__(256)
void convert_T_kernel(const float* __restrict__ Tf, unsigned short* __restrict__ Tb) {
  size_t i = ((size_t)blockIdx.x * 256 + threadIdx.x) * 8;
  float4 a = *(const float4*)(Tf + i);
  float4 b = *(const float4*)(Tf + i + 4);
  union { unsigned short s[8]; uint4 q; } o;
  o.s[0] = f2bf(a.x); o.s[1] = f2bf(a.y); o.s[2] = f2bf(a.z); o.s[3] = f2bf(a.w);
  o.s[4] = f2bf(b.x); o.s[5] = f2bf(b.y); o.s[6] = f2bf(b.z); o.s[7] = f2bf(b.w);
  *(uint4*)(Tb + i) = o.q;
}

// ---------------------------------------------------------------------------
// One recurrence step: x_out = T[act] @ x_in + b[act]; block partial of
// dot(R[act], x_out) -> partials_out[block].  Block 0 additionally folds the
// PREVIOUS step's partials into out[t-1] (stream order guarantees they are
// complete), removing a separate per-step reduce launch.
//
// WMMA mapping (v_wmma_f32_16x16x32_bf16, D = A(16x32) * B(32x16) + C):
//   A[m][k] = x[kbase+k]   broadcast over m, served from LDS
//   B[k][n] = T[row0+n][kbase+k]  per-lane contiguous 32B global loads
//   => D[m][n] = partial y[row0+n] (identical over m); take D row 0.
// ---------------------------------------------------------------------------
template <bool PRECONV>
__global__ __launch_bounds__(BLOCK_THREADS)
void gemv_step_kernel(const unsigned short* __restrict__ Tb,
                      const float* __restrict__ Tf,
                      const float* __restrict__ x_in,
                      float* __restrict__ x_out,
                      const float* __restrict__ b_vecs,
                      const float* __restrict__ R_vecs,
                      const int* __restrict__ actions,
                      const float* __restrict__ partials_prev,
                      float* __restrict__ partials_out,
                      float* __restrict__ out,
                      int t) {
  __shared__ __align__(16) unsigned short xlds[NDIM];   // 8 KB
  __shared__ float ypart[KSPLIT][TILE_ROWS];

  const int tid  = threadIdx.x;
  const int lane = tid & 31;
  const int wave = tid >> 5;
  const int act  = actions[t];                           // uniform scalar load

  // Block 0, wave 0: finalize previous step's reward (overlaps with staging)
  if (t > 0 && blockIdx.x == 0 && tid < 32) {
    float s = 0.0f;
    #pragma unroll
    for (int i = 0; i < NBLOCKS / 32; ++i) s += partials_prev[tid + 32 * i];
    s += __shfl_xor(s, 16, 32);
    s += __shfl_xor(s, 8, 32);
    s += __shfl_xor(s, 4, 32);
    s += __shfl_xor(s, 2, 32);
    s += __shfl_xor(s, 1, 32);
    if (tid == 0) out[t - 1] = s;
  }

  // Stage x (f32 -> bf16) into LDS, fully convergent
  for (int i = tid; i < NDIM / 8; i += BLOCK_THREADS) {
    float4 a = ((const float4*)x_in)[2 * i];
    float4 b = ((const float4*)x_in)[2 * i + 1];
    union { unsigned short s[8]; uint4 q; } o;
    o.s[0] = f2bf(a.x); o.s[1] = f2bf(a.y); o.s[2] = f2bf(a.z); o.s[3] = f2bf(a.w);
    o.s[4] = f2bf(b.x); o.s[5] = f2bf(b.y); o.s[6] = f2bf(b.z); o.s[7] = f2bf(b.w);
    *(uint4*)&xlds[i * 8] = o.q;
  }
  __syncthreads();

  const int half  = lane >> 4;      // which K-half of the 32-chunk this lane holds
  const int rlane = lane & 15;      // N index -> T row within tile
  const int row   = blockIdx.x * TILE_ROWS + rlane;
  const int k0    = wave * KCHUNK;  // this wave's K range

  const size_t rowOff = (size_t)act * NDIM * NDIM + (size_t)row * NDIM
                      + (size_t)(k0 + half * 16);
  const unsigned short* tb = PRECONV ? (Tb + rowOff) : nullptr;
  const float*          tf = PRECONV ? nullptr : (Tf + rowOff);

  v8f acc = {};
  #pragma unroll 4
  for (int kk = 0; kk < KCHUNK; kk += 32) {
    ABReg A, B;
    // A: broadcast x chunk from LDS (two ds_load_b128, conflict-free broadcast)
    const uint4* xa = (const uint4*)&xlds[k0 + kk + half * 16];
    A.q[0] = xa[0];   // K = kk + half*16     .. +8
    A.q[1] = xa[1];   // K = kk + half*16 + 8 .. +8
    if (PRECONV) {
      // B: 16 contiguous bf16 of this lane's T row (two global_load_b128)
      const uint4* tq = (const uint4*)(tb + kk);
      B.q[0] = tq[0];
      B.q[1] = tq[1];
    } else {
      // fallback: read fp32 T and convert in-register
      float4 f0 = *(const float4*)(tf + kk + 0);
      float4 f1 = *(const float4*)(tf + kk + 4);
      float4 f2 = *(const float4*)(tf + kk + 8);
      float4 f3 = *(const float4*)(tf + kk + 12);
      B.us[0]  = f2bf(f0.x); B.us[1]  = f2bf(f0.y); B.us[2]  = f2bf(f0.z); B.us[3]  = f2bf(f0.w);
      B.us[4]  = f2bf(f1.x); B.us[5]  = f2bf(f1.y); B.us[6]  = f2bf(f1.z); B.us[7]  = f2bf(f1.w);
      B.us[8]  = f2bf(f2.x); B.us[9]  = f2bf(f2.y); B.us[10] = f2bf(f2.z); B.us[11] = f2bf(f2.w);
      B.us[12] = f2bf(f3.x); B.us[13] = f2bf(f3.y); B.us[14] = f2bf(f3.z); B.us[15] = f2bf(f3.w);
    }
    // (neg_a, A, neg_b, B, c_mod, C, reuse_a, reuse_b)
    acc = __builtin_amdgcn_wmma_f32_16x16x32_bf16(false, A.v, false, B.v,
                                                  (short)0, acc, false, false);
  }

  // D[0][n] lives in acc[0], lanes 0-15 (all M rows identical by construction)
  if (lane < 16) ypart[wave][lane] = acc[0];
  __syncthreads();

  // Wave 0: combine K-split partials, apply bias, store, reward partial
  if (tid < 32) {
    float pr = 0.0f;
    if (tid < 16) {
      float y = 0.0f;
      #pragma unroll
      for (int w = 0; w < KSPLIT; ++w) y += ypart[w][tid];
      const int r = blockIdx.x * TILE_ROWS + tid;
      y += b_vecs[(size_t)act * NDIM + r];
      x_out[r] = y;
      pr = y * R_vecs[(size_t)act * NDIM + r];
    }
    pr += __shfl_xor(pr, 16, 32);
    pr += __shfl_xor(pr, 8, 32);
    pr += __shfl_xor(pr, 4, 32);
    pr += __shfl_xor(pr, 2, 32);
    pr += __shfl_xor(pr, 1, 32);
    if (tid == 0) partials_out[blockIdx.x] = pr;
  }
}

// Deterministic fixed-tree reduction for the final step's partials
__global__ __launch_bounds__(32)
void reduce_step_kernel(const float* __restrict__ partials, float* __restrict__ out, int t) {
  const int lane = threadIdx.x;
  float s = 0.0f;
  #pragma unroll
  for (int i = 0; i < NBLOCKS / 32; ++i) s += partials[lane + 32 * i];
  s += __shfl_xor(s, 16, 32);
  s += __shfl_xor(s, 8, 32);
  s += __shfl_xor(s, 4, 32);
  s += __shfl_xor(s, 2, 32);
  s += __shfl_xor(s, 1, 32);
  if (lane == 0) out[t] = s;
}

extern "C" void kernel_launch(void* const* d_in, const int* in_sizes, int n_in,
                              void* d_out, int out_size, void* d_ws, size_t ws_size,
                              hipStream_t stream) {
  // setup_inputs order: state, actions, T_mats, b_vecs, R_vecs
  // (state is unused: reference scans from a zero init vector)
  const int*   actions = (const int*)d_in[1];
  const float* T_mats  = (const float*)d_in[2];
  const float* b_vecs  = (const float*)d_in[3];
  const float* R_vecs  = (const float*)d_in[4];
  float* out = (float*)d_out;

  const size_t t_elems = (size_t)N_ACTIONS * NDIM * NDIM;       // 64Mi
  const size_t t_bytes = t_elems * sizeof(unsigned short);      // 128 MiB
  const size_t small   = 2 * NDIM * sizeof(float) + 2 * NBLOCKS * sizeof(float) + 256;

  char* ws = (char*)d_ws;
  size_t off = 0;
  const bool preconv = (ws_size >= t_bytes + small);
  unsigned short* Tb = nullptr;
  if (preconv) { Tb = (unsigned short*)(ws + off); off += t_bytes; }
  float* if0 = (float*)(ws + off); off += NDIM * sizeof(float);
  float* if1 = (float*)(ws + off); off += NDIM * sizeof(float);
  float* pA  = (float*)(ws + off); off += NBLOCKS * sizeof(float);
  float* pB  = (float*)(ws + off); off += NBLOCKS * sizeof(float);

  if (preconv) {
    const int thr = 256;
    const int blocks = (int)(t_elems / ((size_t)thr * 8));      // 32768
    convert_T_kernel<<<blocks, thr, 0, stream>>>(T_mats, Tb);
  }
  hipMemsetAsync(if0, 0, NDIM * sizeof(float), stream);         // x_0 = zeros

  for (int t = 0; t < HORIZON; ++t) {
    const float* xin   = (t & 1) ? if1 : if0;
    float*       xout  = (t & 1) ? if0 : if1;
    float*       pout  = (t & 1) ? pB  : pA;
    const float* pprev = (t & 1) ? pA  : pB;   // written by step t-1
    if (preconv) {
      gemv_step_kernel<true><<<NBLOCKS, BLOCK_THREADS, 0, stream>>>(
          Tb, nullptr, xin, xout, b_vecs, R_vecs, actions, pprev, pout, out, t);
    } else {
      gemv_step_kernel<false><<<NBLOCKS, BLOCK_THREADS, 0, stream>>>(
          nullptr, T_mats, xin, xout, b_vecs, R_vecs, actions, pprev, pout, out, t);
    }
  }
  // Final step's reward partials
  reduce_step_kernel<<<1, 32, 0, stream>>>(
      ((HORIZON - 1) & 1) ? pB : pA, out, HORIZON - 1);
}
